// CrossAttention_30434138259521
// MI455X (gfx1250) — matrix operand order repaired and checked by reading, not compiled, not run on hardware
//
#include <hip/hip_runtime.h>
#include <hip/hip_bf16.h>

// ---------------------------------------------------------------------------
// CDNA5 (gfx1250, wave32) fused attention:
//   1) qkv_proj : X[8192,512]fp32 @ {Wq,Wk,Wv} -> Q/K/V bf16 [B,H,S,Dh]
//   2) attn     : flash attention with v_wmma_f32_16x16x32_bf16, online softmax
//   3) out_proj : O bf16 [8192,512] @ Wo + bo -> fp32 out
// All matmuls run on WMMA bf16 (f32 accumulate). B=4,S=2048,D=512,H=8,Dh=64.
// ---------------------------------------------------------------------------

typedef __bf16 bf16;
typedef __attribute__((ext_vector_type(16))) bf16  v16bf;
typedef __attribute__((ext_vector_type(8)))  bf16  v8bf;
typedef __attribute__((ext_vector_type(4)))  bf16  v4bf;
typedef __attribute__((ext_vector_type(8)))  float v8f;
typedef __attribute__((ext_vector_type(4)))  float v4f;

#define BATCH  4
#define SEQ    2048
#define DMODEL 512
#define NHEADS 8
#define DHEAD  64
#define ATT_SCALE 0.125f   // 1/sqrt(64)

static __device__ __forceinline__ v8f wmma_bf16(v16bf a, v16bf b, v8f c) {
  // (neg_a, A, neg_b, B, c_mod, C, reuse_a, reuse_b)
  return __builtin_amdgcn_wmma_f32_16x16x32_bf16(false, a, false, b, (short)0, c,
                                                 false, false);
}

// A fragment 16x32 (MxK), bf16, from LDS tile (row-major, `pitch` elems/row).
// ISA: lanes 0-15 -> M=lane, VGPR0..3 K=0..7, VGPR4..7 K=16..23
//      lanes 16-31 -> M=lane-16, K offset +8 in each group.
static __device__ __forceinline__ v16bf frag_a(const bf16* lds, int pitch,
                                               int row0, int k0, int lane) {
  const int half = lane >> 4;
  const int m    = lane & 15;
  const bf16* p = lds + (row0 + m) * pitch + k0 + 8 * half;
  v8bf lo = *(const v8bf*)p;          // K = k0 + 8*half + 0..7
  v8bf hi = *(const v8bf*)(p + 16);   // K = k0 + 16 + 8*half + 0..7
  v16bf r;
#pragma unroll
  for (int e = 0; e < 8; ++e) { r[e] = lo[e]; r[e + 8] = hi[e]; }
  return r;
}

// B fragment 32x16 (KxN), bf16. Source LDS tile stored as [n][k] (transposed),
// so lane (half = lane>>4, n = lane&15) reads K = k0 + 8*half + 0..7 and k0 + 16 + 8*half + 0..7.
static __device__ __forceinline__ v16bf frag_b(const bf16* lds, int pitch,
                                               int n0, int k0, int lane) {
  const int half = lane >> 4;
  const int n    = lane & 15;
  const bf16* p = lds + (n0 + n) * pitch + k0 + 8 * half;
  v8bf lo = *(const v8bf*)p;
  v8bf hi = *(const v8bf*)(p + 16);
  v16bf r;
#pragma unroll
  for (int e = 0; e < 8; ++e) { r[e] = lo[e]; r[e + 8] = hi[e]; }
  return r;
}

// ---------------------------------------------------------------------------
// Kernel 1: QKV projection. grid = (64, 4, 3); block = 128 (4 waves).
// Workgroup tile 128x128, wave tile 64x64 (4x4 WMMA accumulators), K step 32.
// ---------------------------------------------------------------------------
__global__ __launch_bounds__(128) void qkv_proj(
    const float* __restrict__ X,
    const float* __restrict__ Wq, const float* __restrict__ Wk,
    const float* __restrict__ Wv,
    bf16* __restrict__ Qw, bf16* __restrict__ Kw, bf16* __restrict__ Vw) {
  __shared__ __align__(16) bf16 As[128][40];  // X tile, row-major [m][k]
  __shared__ __align__(16) bf16 Bs[128][40];  // W tile, transposed [n][k]

  const int tid  = threadIdx.x;
  const int lane = tid & 31;
  const int wid  = tid >> 5;
  const int m0   = blockIdx.x * 128;
  const int n0   = blockIdx.y * 128;
  const float* W = (blockIdx.z == 0) ? Wq : (blockIdx.z == 1 ? Wk : Wv);
  bf16* Out      = (blockIdx.z == 0) ? Qw : (blockIdx.z == 1 ? Kw : Vw);

  const int wm = (wid >> 1) * 64;
  const int wn = (wid & 1) * 64;

  v8f acc[4][4];
#pragma unroll
  for (int mi = 0; mi < 4; ++mi)
#pragma unroll
    for (int ni = 0; ni < 4; ++ni)
#pragma unroll
      for (int e = 0; e < 8; ++e) acc[mi][ni][e] = 0.f;

  for (int kt = 0; kt < DMODEL / 32; ++kt) {
    const int kg = kt * 32;
    // Stage X tile 128x32 fp32 -> bf16 LDS (coalesced float4 reads)
#pragma unroll
    for (int it = 0; it < 8; ++it) {
      int i  = tid + it * 128;        // 1024 float4-chunks
      int m  = i >> 3;
      int kq = (i & 7) * 4;
      v4f xv = *(const v4f*)(X + (size_t)(m0 + m) * DMODEL + kg + kq);
      v4bf bv;
      bv[0] = (bf16)xv[0]; bv[1] = (bf16)xv[1];
      bv[2] = (bf16)xv[2]; bv[3] = (bf16)xv[3];
      *(v4bf*)&As[m][kq] = bv;
    }
    // Stage W tile 32x128 transposed: thread t owns output column n0+t
    {
      const int n = n0 + tid;
#pragma unroll
      for (int c = 0; c < 4; ++c) {
        v8bf t;
#pragma unroll
        for (int e = 0; e < 8; ++e)
          t[e] = (bf16)W[(size_t)(kg + c * 8 + e) * DMODEL + n];
        *(v8bf*)&Bs[tid][c * 8] = t;
      }
    }
    __syncthreads();
#pragma unroll
    for (int mi = 0; mi < 4; ++mi) {
      v16bf a = frag_a(&As[0][0], 40, wm + 16 * mi, 0, lane);
#pragma unroll
      for (int ni = 0; ni < 4; ++ni) {
        v16bf bb = frag_b(&Bs[0][0], 40, wn + 16 * ni, 0, lane);
        acc[mi][ni] = wmma_bf16(a, bb, acc[mi][ni]);
      }
    }
    __syncthreads();
  }

  // Store bf16 into head-major layout [B,H,S,Dh]
  const int half = lane >> 4;
  const int nn   = lane & 15;
#pragma unroll
  for (int mi = 0; mi < 4; ++mi)
#pragma unroll
    for (int ni = 0; ni < 4; ++ni) {
      const int col = n0 + wn + 16 * ni + nn;
      const int hh = col >> 6, dh = col & 63;
#pragma unroll
      for (int j = 0; j < 8; ++j) {
        const int r = m0 + wm + 16 * mi + 8 * half + j;  // C layout: M=j+8*half
        const int bb = r >> 11, s = r & 2047;
        Out[((size_t)(bb * NHEADS + hh) * SEQ + s) * DHEAD + dh] =
            (bf16)acc[mi][ni][j];
      }
    }
}

// ---------------------------------------------------------------------------
// Kernel 2: flash attention. grid = (S/64, H, B); block = 128 (4 waves).
// Each wave owns 16 query rows; key loop in chunks of 64.
// ---------------------------------------------------------------------------
__global__ __launch_bounds__(128) void attn(
    const bf16* __restrict__ Qw, const bf16* __restrict__ Kw,
    const bf16* __restrict__ Vw, bf16* __restrict__ Ow) {
  __shared__ __align__(16) bf16 Qs[64][72];  // [q][dh]
  __shared__ __align__(16) bf16 Ks[64][72];  // [sk][dh]
  __shared__ __align__(16) bf16 Vt[64][72];  // [dh][sk] (transposed)
  __shared__ __align__(16) bf16 Ps[64][72];  // P tiles, wave w owns rows 16w..

  const int tid  = threadIdx.x;
  const int lane = tid & 31;
  const int wid  = tid >> 5;
  const int q0   = blockIdx.x * 64;
  const int hh   = blockIdx.y;
  const int bb   = blockIdx.z;
  const size_t base = ((size_t)bb * NHEADS + hh) * (size_t)SEQ * DHEAD;

  // Load 64x64 Q block
#pragma unroll
  for (int it = 0; it < 4; ++it) {
    int i = tid + it * 128;
    int row = i >> 3, c8 = (i & 7) * 8;
    *(v8bf*)&Qs[row][c8] =
        *(const v8bf*)(Qw + base + (size_t)(q0 + row) * DHEAD + c8);
  }
  __syncthreads();

  const v16bf qa0 = frag_a(&Qs[0][0], 72, wid * 16, 0, lane);
  const v16bf qa1 = frag_a(&Qs[0][0], 72, wid * 16, 32, lane);

  v8f o[4];
  float mrow[8], lrow[8];
#pragma unroll
  for (int ni = 0; ni < 4; ++ni)
#pragma unroll
    for (int e = 0; e < 8; ++e) o[ni][e] = 0.f;
#pragma unroll
  for (int j = 0; j < 8; ++j) { mrow[j] = -1e30f; lrow[j] = 0.f; }

  for (int kc = 0; kc < SEQ / 64; ++kc) {
    const int s0 = kc * 64;
    // Stage K rows (row-major) and V transposed
#pragma unroll
    for (int it = 0; it < 4; ++it) {
      int i = tid + it * 128;
      int row = i >> 3, c8 = (i & 7) * 8;
      *(v8bf*)&Ks[row][c8] =
          *(const v8bf*)(Kw + base + (size_t)(s0 + row) * DHEAD + c8);
    }
#pragma unroll
    for (int it = 0; it < 8; ++it) {
      int i = tid + it * 128;
      int row = i >> 4, c4 = (i & 15) * 4;
      v4bf v = *(const v4bf*)(Vw + base + (size_t)(s0 + row) * DHEAD + c4);
#pragma unroll
      for (int e = 0; e < 4; ++e) Vt[c4 + e][row] = v[e];
    }
    __syncthreads();

    // Prefetch next chunk while this one computes (global_prefetch_b8)
    if (kc + 1 < SEQ / 64) {
      const size_t nr = base + (size_t)(s0 + 64 + (tid >> 1)) * DHEAD +
                        (size_t)(tid & 1) * 32;
      __builtin_prefetch(Kw + nr, 0, 1);
      __builtin_prefetch(Vw + nr, 0, 1);
    }

    // S = Q Kt  (B-fragment rows are key indices; Ks row-major works directly)
    v8f sc[4];
#pragma unroll
    for (int ni = 0; ni < 4; ++ni) {
#pragma unroll
      for (int e = 0; e < 8; ++e) sc[ni][e] = 0.f;
      v16bf b0 = frag_b(&Ks[0][0], 72, 16 * ni, 0, lane);
      sc[ni] = wmma_bf16(qa0, b0, sc[ni]);
      v16bf b1 = frag_b(&Ks[0][0], 72, 16 * ni, 32, lane);
      sc[ni] = wmma_bf16(qa1, b1, sc[ni]);
    }

    // Online softmax. C layout: row M = j + 8*half, col N = lane&15.
    {
#pragma unroll
      for (int j = 0; j < 8; ++j) {
        float pm = -1e30f;
#pragma unroll
        for (int ni = 0; ni < 4; ++ni) {
          sc[ni][j] *= ATT_SCALE;
          pm = fmaxf(pm, sc[ni][j]);
        }
#pragma unroll
        for (int off = 1; off <= 8; off <<= 1)  // reduce across 16-lane group
          pm = fmaxf(pm, __shfl_xor(pm, off, 32));
        const float nm  = fmaxf(mrow[j], pm);
        const float fac = __expf(mrow[j] - nm);
        mrow[j] = nm;
        float rs = 0.f;
#pragma unroll
        for (int ni = 0; ni < 4; ++ni) {
          const float p = __expf(sc[ni][j] - nm);
          sc[ni][j] = p;
          rs += p;
        }
#pragma unroll
        for (int off = 1; off <= 8; off <<= 1)
          rs += __shfl_xor(rs, off, 32);
        lrow[j] = lrow[j] * fac + rs;
#pragma unroll
        for (int ni = 0; ni < 4; ++ni) o[ni][j] *= fac;
      }
      // C-layout -> LDS so next WMMA can read P as an A fragment
      const int half = lane >> 4, nn = lane & 15;
#pragma unroll
      for (int ni = 0; ni < 4; ++ni)
#pragma unroll
        for (int j = 0; j < 8; ++j)
          Ps[wid * 16 + 8 * half + j][16 * ni + nn] = (bf16)sc[ni][j];
    }
    __syncthreads();

    // O += P @ V  (Vt is [dh][sk] so B fragments are contiguous)
#pragma unroll
    for (int kk = 0; kk < 2; ++kk) {
      v16bf pa = frag_a(&Ps[0][0], 72, wid * 16, kk * 32, lane);
#pragma unroll
      for (int ni = 0; ni < 4; ++ni) {
        v16bf vb = frag_b(&Vt[0][0], 72, 16 * ni, kk * 32, lane);
        o[ni] = wmma_bf16(pa, vb, o[ni]);
      }
    }
    __syncthreads();
  }

  // Normalize and store O as bf16 in [B,S,D] for the output projection
  const int half = lane >> 4, nn = lane & 15;
#pragma unroll
  for (int j = 0; j < 8; ++j) {
    const float inv = 1.f / lrow[j];
    const int s = q0 + wid * 16 + 8 * half + j;
#pragma unroll
    for (int ni = 0; ni < 4; ++ni) {
      const int d = hh * DHEAD + 16 * ni + nn;
      Ow[((size_t)bb * SEQ + s) * DMODEL + d] = (bf16)(o[ni][j] * inv);
    }
  }
}

// ---------------------------------------------------------------------------
// Kernel 3: output projection, fp32 result + bias. grid = (64, 4); block 128.
// ---------------------------------------------------------------------------
__global__ __launch_bounds__(128) void out_proj(
    const bf16* __restrict__ Ain, const float* __restrict__ Wo,
    const float* __restrict__ bo, float* __restrict__ Y) {
  __shared__ __align__(16) bf16 As[128][40];
  __shared__ __align__(16) bf16 Bs[128][40];

  const int tid  = threadIdx.x;
  const int lane = tid & 31;
  const int wid  = tid >> 5;
  const int m0   = blockIdx.x * 128;
  const int n0   = blockIdx.y * 128;
  const int wm   = (wid >> 1) * 64;
  const int wn   = (wid & 1) * 64;

  v8f acc[4][4];
#pragma unroll
  for (int mi = 0; mi < 4; ++mi)
#pragma unroll
    for (int ni = 0; ni < 4; ++ni)
#pragma unroll
      for (int e = 0; e < 8; ++e) acc[mi][ni][e] = 0.f;

  for (int kt = 0; kt < DMODEL / 32; ++kt) {
    const int kg = kt * 32;
#pragma unroll
    for (int it = 0; it < 4; ++it) {
      int i = tid + it * 128;
      int m = i >> 2, c8 = (i & 3) * 8;
      *(v8bf*)&As[m][c8] =
          *(const v8bf*)(Ain + (size_t)(m0 + m) * DMODEL + kg + c8);
    }
    {
      const int n = n0 + tid;
#pragma unroll
      for (int c = 0; c < 4; ++c) {
        v8bf t;
#pragma unroll
        for (int e = 0; e < 8; ++e)
          t[e] = (bf16)Wo[(size_t)(kg + c * 8 + e) * DMODEL + n];
        *(v8bf*)&Bs[tid][c * 8] = t;
      }
    }
    __syncthreads();
#pragma unroll
    for (int mi = 0; mi < 4; ++mi) {
      v16bf a = frag_a(&As[0][0], 40, wm + 16 * mi, 0, lane);
#pragma unroll
      for (int ni = 0; ni < 4; ++ni) {
        v16bf bb = frag_b(&Bs[0][0], 40, wn + 16 * ni, 0, lane);
        acc[mi][ni] = wmma_bf16(a, bb, acc[mi][ni]);
      }
    }
    __syncthreads();
  }

  const int half = lane >> 4, nn = lane & 15;
#pragma unroll
  for (int mi = 0; mi < 4; ++mi)
#pragma unroll
    for (int ni = 0; ni < 4; ++ni) {
      const int col = n0 + wn + 16 * ni + nn;
      const float bias = bo[col];
#pragma unroll
      for (int j = 0; j < 8; ++j) {
        const int r = m0 + wm + 16 * mi + 8 * half + j;
        Y[(size_t)r * DMODEL + col] = acc[mi][ni][j] + bias;
      }
    }
}

// ---------------------------------------------------------------------------
extern "C" void kernel_launch(void* const* d_in, const int* in_sizes, int n_in,
                              void* d_out, int out_size, void* d_ws,
                              size_t ws_size, hipStream_t stream) {
  (void)in_sizes; (void)n_in; (void)out_size; (void)ws_size;
  const float* X  = (const float*)d_in[0];
  const float* Wq = (const float*)d_in[1];
  const float* Wk = (const float*)d_in[2];
  const float* Wv = (const float*)d_in[3];
  const float* Wo = (const float*)d_in[4];
  const float* bo = (const float*)d_in[5];
  float* Y = (float*)d_out;

  const size_t elems = (size_t)BATCH * SEQ * DMODEL;  // 4,194,304
  bf16* Qw = (bf16*)d_ws;          // 8 MB each; total ws use = 32 MB
  bf16* Kw = Qw + elems;
  bf16* Vw = Kw + elems;
  bf16* Ow = Vw + elems;

  dim3 g1((BATCH * SEQ) / 128, DMODEL / 128, 3);
  qkv_proj<<<g1, 128, 0, stream>>>(X, Wq, Wk, Wv, Qw, Kw, Vw);

  dim3 g2(SEQ / 64, NHEADS, BATCH);
  attn<<<g2, 128, 0, stream>>>(Qw, Kw, Vw, Ow);

  dim3 g3((BATCH * SEQ) / 128, DMODEL / 128);
  out_proj<<<g3, 128, 0, stream>>>(Ow, Wo, bo, Y);
}
